// MultiheadAttentionSuper_42374147342405
// MI455X (gfx1250) — matrix operand order repaired and checked by reading, not compiled, not run on hardware
//
#include <hip/hip_runtime.h>
#include <hip/hip_bf16.h>

// ---------------------------------------------------------------------------
// MHA forward for MI455X (gfx1250): bf16 WMMA (v_wmma_f32_16x16x32_bf16),
// fp32 accumulation, async global->LDS staging (GLOBAL_LOAD_ASYNC_TO_LDS_B128).
// T=2048 B=2 E=1024 H=16 DH=64.
// ---------------------------------------------------------------------------

typedef __bf16 bf16_t;
typedef __attribute__((ext_vector_type(8)))  __bf16 v8bf;
typedef __attribute__((ext_vector_type(16))) __bf16 v16bf;
typedef __attribute__((ext_vector_type(8)))  float  v8f;
typedef __attribute__((ext_vector_type(4)))  int    v4i;
typedef __attribute__((address_space(1))) v4i gas_v4i;   // global-space v4i
typedef __attribute__((address_space(3))) v4i las_v4i;   // LDS-space v4i

#define T_DIM 2048
#define B_DIM 2
#define E_DIM 1024
#define H_DIM 16
#define DH    64
#define M_ROWS (T_DIM * B_DIM)   // 4096
#define N_QKV  (3 * E_DIM)       // 3072

// float -> bf16, round-to-nearest-even via bit ops
static __device__ __forceinline__ bf16_t f2bf(float x) {
  union { float f; unsigned u; } v; v.f = x;
  unsigned r = v.u + 0x7FFFu + ((v.u >> 16) & 1u);
  union { unsigned short s; bf16_t b; } o;
  o.s = (unsigned short)(r >> 16);
  return o.b;
}

// build one 16-wide bf16 WMMA operand from two 16-byte loads
static __device__ __forceinline__ v16bf load2(const bf16_t* p0, const bf16_t* p1) {
  v8bf a = *(const v8bf*)p0;
  v8bf b = *(const v8bf*)p1;
  v16bf r;
#pragma unroll
  for (int i = 0; i < 8; ++i) { r[i] = a[i]; r[i + 8] = b[i]; }
  return r;
}

static __device__ __forceinline__ v8f wmma_bf16(v16bf a, v16bf b, v8f c) {
  return __builtin_amdgcn_wmma_f32_16x16x32_bf16(false, a, false, b,
                                                 (short)0, c, false, false);
}

// async 16B global -> LDS (GLOBAL_LOAD_ASYNC_TO_LDS_B128, ASYNCcnt-tracked);
// falls back to a VGPR round-trip if the builtin is unavailable.
static __device__ __forceinline__ void async_ld16(const bf16_t* g, bf16_t* l) {
#if __has_builtin(__builtin_amdgcn_global_load_async_to_lds_b128)
  __builtin_amdgcn_global_load_async_to_lds_b128(
      (gas_v4i*)g, (las_v4i*)l, 0, 0);
#else
  *(v8bf*)l = *(const v8bf*)g;
#endif
}

static __device__ __forceinline__ void wait_async0() {
#if __has_builtin(__builtin_amdgcn_global_load_async_to_lds_b128)
#if __has_builtin(__builtin_amdgcn_s_wait_asynccnt)
  __builtin_amdgcn_s_wait_asynccnt(0);
#else
  asm volatile("s_wait_asynccnt 0x0" ::: "memory");
#endif
#endif
}

// ---------------------------------------------------------------------------
// fp32 -> bf16 conversion (grid-stride)
// ---------------------------------------------------------------------------
__global__ void cvt_f32_bf16(const float* __restrict__ in,
                             bf16_t* __restrict__ out, int n) {
  int i = blockIdx.x * blockDim.x + threadIdx.x;
  int stride = gridDim.x * blockDim.x;
  for (; i < n; i += stride) out[i] = f2bf(in[i]);
}

// ---------------------------------------------------------------------------
// Tiled GEMM:  C[m,n] = sum_k A[m,k] * W[n,k]   (A: MxK, W: NxK, row-major)
// Block tile 128x128, 256 threads = 8 waves (4 M-waves x 2 N-waves),
// wave tile 32x64 -> 2x4 accumulator tiles, K step 32, async LDS staging.
// MODE 0: QKV epilogue (+bias, Q*0.125, scatter to Q/K/V^T head layouts, bf16)
// MODE 1: out-proj epilogue (+bias, fp32 to d_out)
// ---------------------------------------------------------------------------
template <int MODE>
__global__ __launch_bounds__(256)
void gemm_nt(const bf16_t* __restrict__ A, const bf16_t* __restrict__ W,
             const float* __restrict__ bias, float* __restrict__ outF,
             bf16_t* __restrict__ Qb, bf16_t* __restrict__ Kb,
             bf16_t* __restrict__ Vt, int Kdim) {
  __shared__ __align__(16) bf16_t As[128][40];   // 80B row pitch (16B aligned)
  __shared__ __align__(16) bf16_t Bs[128][40];

  const int tid   = threadIdx.x;
  const int lane  = tid & 31;
  const int wave  = tid >> 5;
  const int wm    = wave >> 1;          // 0..3
  const int wn    = wave & 1;           // 0..1
  const int m0    = blockIdx.x * 128;
  const int n0    = blockIdx.y * 128;
  const int lhalf = (lane < 16) ? 0 : 1;
  const int lmod  = lane & 15;

  v8f acc[2][4];
#pragma unroll
  for (int mi = 0; mi < 2; ++mi)
#pragma unroll
    for (int ni = 0; ni < 4; ++ni)
#pragma unroll
      for (int e = 0; e < 8; ++e) acc[mi][ni][e] = 0.0f;

  for (int k0 = 0; k0 < Kdim; k0 += 32) {
    // stage 128x32 bf16 tiles of A and W into LDS (async b128, 4/thread)
#pragma unroll
    for (int t2 = 0; t2 < 2; ++t2) {
      int idx = tid + t2 * 256;        // 0..511
      int r   = idx >> 2;              // 0..127
      int c   = (idx & 3) * 8;         // 0,8,16,24
      async_ld16(&A[(size_t)(m0 + r) * Kdim + k0 + c], &As[r][c]);
      async_ld16(&W[(size_t)(n0 + r) * Kdim + k0 + c], &Bs[r][c]);
    }
    wait_async0();
    __syncthreads();

    // A fragments: lane<16 -> K {0..7,16..23}; lane>=16 -> K {8..15,24..31}
    v16bf aF[2];
#pragma unroll
    for (int mi = 0; mi < 2; ++mi) {
      int r = wm * 32 + mi * 16 + lmod;
      aF[mi] = load2(&As[r][lhalf * 8], &As[r][16 + lhalf * 8]);
    }
    // B fragments: lane holds column lmod; lane<16 -> K {0..7,16..23}; lane>=16 -> K {8..15,24..31}
    v16bf bF[4];
#pragma unroll
    for (int ni = 0; ni < 4; ++ni) {
      int r = wn * 64 + ni * 16 + lmod;
      bF[ni] = load2(&Bs[r][lhalf * 8], &Bs[r][16 + lhalf * 8]);
    }

#pragma unroll
    for (int mi = 0; mi < 2; ++mi)
#pragma unroll
      for (int ni = 0; ni < 4; ++ni)
        acc[mi][ni] = wmma_bf16(aF[mi], bF[ni], acc[mi][ni]);
    __syncthreads();
  }

  // epilogue: C-layout row = r + 8*(lane>=16), col = lane%16
#pragma unroll
  for (int mi = 0; mi < 2; ++mi) {
#pragma unroll
    for (int ni = 0; ni < 4; ++ni) {
      const int gn = n0 + wn * 64 + ni * 16 + lmod;
      const float bv = bias[gn];
#pragma unroll
      for (int r = 0; r < 8; ++r) {
        const int gm = m0 + wm * 32 + mi * 16 + r + lhalf * 8;
        float v = acc[mi][ni][r] + bv;
        if (MODE == 0) {
          const int which = gn >> 10;          // 0:Q 1:K 2:V
          const int e2 = gn & 1023;
          const int h = e2 >> 6, d = e2 & 63;
          const int t = gm >> 1, b = gm & 1;   // m = t*B + b
          const int head = b * H_DIM + h;
          if (which == 0) {
            Qb[((size_t)head * T_DIM + t) * DH + d] = f2bf(v * 0.125f);
          } else if (which == 1) {
            Kb[((size_t)head * T_DIM + t) * DH + d] = f2bf(v);
          } else {  // V stored transposed: [head][d][t]
            Vt[((size_t)head * DH + d) * T_DIM + t] = f2bf(v);
          }
        } else {
          outF[(size_t)gm * E_DIM + gn] = v;   // [T,B,E] flat
        }
      }
    }
  }
}

// ---------------------------------------------------------------------------
// Flash attention per head.  grid = (T/64, B*H), 128 threads = 4 waves,
// each wave owns 16 query rows; keys processed in tiles of 64.
// Row-sums of P computed by WMMA against an all-ones B operand (no shuffle
// reduction); only the row-max needs a 4-step ds_bpermute butterfly.
// ---------------------------------------------------------------------------
__global__ __launch_bounds__(128)
void attn_kernel(const bf16_t* __restrict__ Qb, const bf16_t* __restrict__ Kb,
                 const bf16_t* __restrict__ Vt, bf16_t* __restrict__ AttnB) {
  __shared__ __align__(16) bf16_t Plds[4][16][72];  // wave-private 16x64 tiles

  const int lane  = threadIdx.x & 31;
  const int wave  = threadIdx.x >> 5;
  const int lhalf = (lane < 16) ? 0 : 1;
  const int lmod  = lane & 15;
  const int head  = blockIdx.y;                 // = b*H + h
  const int t0    = blockIdx.x * 64 + wave * 16;

  const bf16_t* Qh = Qb + (size_t)head * T_DIM * DH;
  const bf16_t* Kh = Kb + (size_t)head * T_DIM * DH;
  const bf16_t* Vh = Vt + (size_t)head * DH * T_DIM;

  // Q fragments (A-layout), resident for the whole kernel
  v16bf qF[2];
  {
    const bf16_t* qr = Qh + (size_t)(t0 + lmod) * DH;
    qF[0] = load2(qr + lhalf * 8,      qr + 16 + lhalf * 8);
    qF[1] = load2(qr + 32 + lhalf * 8, qr + 48 + lhalf * 8);
  }

  // all-ones B operand for row-sum WMMA
  v16bf onesB;
#pragma unroll
  for (int i = 0; i < 16; ++i) onesB[i] = f2bf(1.0f);

  // o[0..3] = output accumulators (16x64), o[4] = softmax denominator rows
  v8f o[5];
#pragma unroll
  for (int n = 0; n < 5; ++n)
#pragma unroll
    for (int e = 0; e < 8; ++e) o[n][e] = 0.0f;
  float m_i[8];
#pragma unroll
  for (int r = 0; r < 8; ++r) m_i[r] = -1e30f;

  for (int kt = 0; kt < T_DIM; kt += 64) {
    // S = Q * K^T over 64 keys: 4 tiles of 16 keys, 2 WMMAs each
    v8f s[4];
#pragma unroll
    for (int j = 0; j < 4; ++j) {
      const bf16_t* kr = Kh + (size_t)(kt + j * 16 + lmod) * DH;
      v16bf kf0 = load2(kr + lhalf * 8,      kr + 16 + lhalf * 8);
      v16bf kf1 = load2(kr + 32 + lhalf * 8, kr + 48 + lhalf * 8);
#pragma unroll
      for (int e = 0; e < 8; ++e) s[j][e] = 0.0f;
      s[j] = wmma_bf16(qF[0], kf0, s[j]);
      s[j] = wmma_bf16(qF[1], kf1, s[j]);
    }

    // online softmax (row = r + 8*lhalf): local max of 4 tiles, one butterfly
#pragma unroll
    for (int r = 0; r < 8; ++r) {
      float mx = fmaxf(fmaxf(s[0][r], s[1][r]), fmaxf(s[2][r], s[3][r]));
#pragma unroll
      for (int off = 1; off < 16; off <<= 1)
        mx = fmaxf(mx, __shfl_xor(mx, off, 32));
      const float mnew = fmaxf(m_i[r], mx);
      const float corr = __expf(m_i[r] - mnew);
      m_i[r] = mnew;
#pragma unroll
      for (int n = 0; n < 5; ++n) o[n][r] *= corr;   // rescale O and denom
#pragma unroll
      for (int j = 0; j < 4; ++j)
        Plds[wave][r + lhalf * 8][j * 16 + lmod] = f2bf(__expf(s[j][r] - mnew));
    }
    asm volatile("s_wait_dscnt 0x0" ::: "memory");

    // reload P in A-layout: two 16x32 fragments (keys 0..31, 32..63)
    const bf16_t* pr = &Plds[wave][lmod][0];
    v16bf pF0 = load2(pr + lhalf * 8,      pr + 16 + lhalf * 8);
    v16bf pF1 = load2(pr + 32 + lhalf * 8, pr + 48 + lhalf * 8);

    // O += P * V  (V^T layout: lane = output d column, keys contiguous)
#pragma unroll
    for (int n = 0; n < 4; ++n) {
      const bf16_t* vp = Vh + (size_t)(n * 16 + lmod) * T_DIM + kt;
      v16bf vF0 = load2(vp + lhalf * 8,      vp + 16 + lhalf * 8);
      v16bf vF1 = load2(vp + 32 + lhalf * 8, vp + 48 + lhalf * 8);
      o[n] = wmma_bf16(pF0, vF0, o[n]);
      o[n] = wmma_bf16(pF1, vF1, o[n]);
    }
    // denominator += row-sum(P) via WMMA with ones
    o[4] = wmma_bf16(pF0, onesB, o[4]);
    o[4] = wmma_bf16(pF1, onesB, o[4]);
  }

  // normalize and write attn output in [M=T*B, E] bf16 for the final GEMM
  const int b = head >> 4;
  const int h = head & 15;
#pragma unroll
  for (int n = 0; n < 4; ++n) {
#pragma unroll
    for (int r = 0; r < 8; ++r) {
      const int t = t0 + r + lhalf * 8;
      const int m = t * B_DIM + b;
      const int col = h * DH + n * 16 + lmod;
      AttnB[(size_t)m * E_DIM + col] = f2bf(o[n][r] / o[4][r]);
    }
  }
}

// ---------------------------------------------------------------------------
extern "C" void kernel_launch(void* const* d_in, const int* in_sizes, int n_in,
                              void* d_out, int out_size, void* d_ws,
                              size_t ws_size, hipStream_t stream) {
  const float* q_in    = (const float*)d_in[0];  // query [T,B,E]
  const float* w_in    = (const float*)d_in[3];  // in_proj_weight [3E,E]
  const float* bias_in = (const float*)d_in[4];  // in_proj_bias [3E]
  const float* ow_in   = (const float*)d_in[5];  // out_w [E,E]
  const float* ob_in   = (const float*)d_in[6];  // out_b [E]
  float* out = (float*)d_out;

  char* ws = (char*)d_ws;
  size_t off = 0;
  bf16_t* Xbf  = (bf16_t*)(ws + off); off += (size_t)M_ROWS * E_DIM * 2;
  bf16_t* Wbf  = (bf16_t*)(ws + off); off += (size_t)N_QKV * E_DIM * 2;
  bf16_t* OWbf = (bf16_t*)(ws + off); off += (size_t)E_DIM * E_DIM * 2;
  bf16_t* Qb   = (bf16_t*)(ws + off); off += (size_t)B_DIM * H_DIM * T_DIM * DH * 2;
  bf16_t* Kb   = (bf16_t*)(ws + off); off += (size_t)B_DIM * H_DIM * T_DIM * DH * 2;
  bf16_t* Vt   = (bf16_t*)(ws + off); off += (size_t)B_DIM * H_DIM * T_DIM * DH * 2;
  bf16_t* AttnB= (bf16_t*)(ws + off); off += (size_t)M_ROWS * E_DIM * 2;

  // 1) fp32 -> bf16 converts
  cvt_f32_bf16<<<2048, 256, 0, stream>>>(q_in,  Xbf,  M_ROWS * E_DIM);
  cvt_f32_bf16<<<2048, 256, 0, stream>>>(w_in,  Wbf,  N_QKV * E_DIM);
  cvt_f32_bf16<<<2048, 256, 0, stream>>>(ow_in, OWbf, E_DIM * E_DIM);

  // 2) QKV projection GEMM: [4096,1024] x [3072,1024]^T
  {
    dim3 grid(M_ROWS / 128, N_QKV / 128);   // 32 x 24
    gemm_nt<0><<<grid, 256, 0, stream>>>(Xbf, Wbf, bias_in, nullptr,
                                         Qb, Kb, Vt, E_DIM);
  }

  // 3) flash attention per head
  {
    dim3 grid(T_DIM / 64, B_DIM * H_DIM);   // 32 x 32
    attn_kernel<<<grid, 128, 0, stream>>>(Qb, Kb, Vt, AttnB);
  }

  // 4) output projection GEMM: [4096,1024] x [1024,1024]^T -> fp32 d_out
  {
    dim3 grid(M_ROWS / 128, E_DIM / 128);   // 32 x 8
    gemm_nt<1><<<grid, 256, 0, stream>>>(AttnB, OWbf, ob_in, out,
                                         nullptr, nullptr, nullptr, E_DIM);
  }
}
